// Memory_39651138077431
// MI455X (gfx1250) — matrix operand order repaired and checked by reading, not compiled, not run on hardware
//
#include <hip/hip_runtime.h>
#include <hip/hip_bf16.h>

typedef __attribute__((ext_vector_type(16))) __bf16 v16bf;
typedef __attribute__((ext_vector_type(8)))  __bf16 v8bf;
typedef __attribute__((ext_vector_type(8)))  float  v8f;

#define T_STEPS 128
#define BATCH   256
#define CIN     1024
#define HDIM    1024
#define FOURH   4096

__device__ __forceinline__ float fast_tanh(float x) {
#if __has_builtin(__builtin_amdgcn_tanhf)
  return __builtin_amdgcn_tanhf(x);
#else
  return tanhf(x);
#endif
}
__device__ __forceinline__ float fast_sigmoid(float x) {
  return 0.5f * fast_tanh(0.5f * x) + 0.5f;
}

// ---- CDNA5 async copy: one 16-byte global->LDS transfer, ASYNCcnt-tracked ----
// VDST holds the wave-relative LDS byte offset (= low 32 bits of a generic
// pointer to __shared__ memory).
__device__ __forceinline__ void async_copy_b128(const __bf16* gp, __bf16* lp) {
  unsigned lds_off = (unsigned)(size_t)(const void*)lp;
  asm volatile("global_load_async_to_lds_b128 %0, %1, off"
               :: "v"(lds_off), "v"(gp) : "memory");
}
__device__ __forceinline__ void wait_async_le1() {
  asm volatile("s_wait_asynccnt 0x1" ::: "memory");
}
__device__ __forceinline__ void wait_async_le0() {
  asm volatile("s_wait_asynccnt 0x0" ::: "memory");
}

// ---- WMMA fragment loaders (bf16, 16x16x32, wave32, contiguous-k order shared by A and B) ----
// A: 16x32 (MxK). Lanes 0-15: row M=lane, K=0..15 contiguous.
//    Lanes 16-31: row M=lane-16, K=16..31 contiguous.
__device__ __forceinline__ v16bf load_a_frag(const __bf16* __restrict__ src, int ld,
                                             int row0, int k0, int lane) {
  int r  = row0 + (lane & 15);
  int kh = k0 + ((lane >> 4) << 4);
  union { v16bf v; v8bf h[2]; } u;
  const __bf16* p = src + (size_t)r * ld + kh;
  u.h[0] = *(const v8bf*)(p);
  u.h[1] = *(const v8bf*)(p + 8);
  return u.v;
}

// B: 32x16 (KxN) from an N-major source: lane n holds K=0..15 (lanes 0-15)
// or K=16..31 (lanes 16-31) of column n, contiguous halves.
__device__ __forceinline__ v16bf load_b_frag(const __bf16* __restrict__ bt, int ld,
                                             int n0, int k0, int lane) {
  int n  = n0 + (lane & 15);
  int kb = k0 + ((lane >> 4) << 4);
  union { v16bf v; v8bf h[2]; } u;
  const __bf16* p = bt + (size_t)n * ld + kb;
  u.h[0] = *(const v8bf*)(p);
  u.h[1] = *(const v8bf*)(p + 8);
  return u.v;
}

// Same B fragment but from an LDS tile laid out as [n][k] with k-extent 32.
__device__ __forceinline__ v16bf load_b_lds(const __bf16* s, int lane) {
  int n  = lane & 15;
  int kb = (lane >> 4) << 4;
  union { v16bf v; v8bf h[2]; } u;
  const __bf16* p = s + n * 32 + kb;
  u.h[0] = *(const v8bf*)(p);
  u.h[1] = *(const v8bf*)(p + 8);
  return u.v;
}

// ---- elementwise prep kernels ----
__global__ __launch_bounds__(256) void k_cast_bf16(const float* __restrict__ src,
                                                   __bf16* __restrict__ dst, int n) {
  int i = blockIdx.x * blockDim.x + threadIdx.x;
  if (i < n) dst[i] = (__bf16)src[i];
}

// src: K x N row-major f32 -> dst: N x K row-major bf16 (transpose + cast)
__global__ __launch_bounds__(256) void k_transpose_bf16(const float* __restrict__ src,
                                                        __bf16* __restrict__ dst,
                                                        int K, int N) {
  int i = blockIdx.x * blockDim.x + threadIdx.x;   // over N*K
  if (i >= N * K) return;
  int n = i / K;
  int k = i - n * K;
  dst[(size_t)n * K + k] = (__bf16)src[(size_t)k * N + n];
}

__global__ __launch_bounds__(256) void k_init_state(const float* __restrict__ c1,
                                                    const float* __restrict__ h1,
                                                    const float* __restrict__ c2,
                                                    const float* __restrict__ h2,
                                                    float* __restrict__ c1s,
                                                    float* __restrict__ c2s,
                                                    __bf16* __restrict__ h1b,
                                                    __bf16* __restrict__ h2b, int n) {
  int i = blockIdx.x * blockDim.x + threadIdx.x;
  if (i >= n) return;
  c1s[i] = c1[i];
  c2s[i] = c2[i];
  h1b[i] = (__bf16)h1[i];
  h2b[i] = (__bf16)h2[i];
}

// ---- big GEMM: Z[32768,4096] = X(bf16) @ Wi + b, f32 accum ----
// One wave computes a 16x64 output block (A-frag reused across 4 N-tiles).
// This kernel is HBM-bound on the Z stream; keep it simple and wide.
__global__ __launch_bounds__(256) void k_gemm_xwi(const __bf16* __restrict__ xb,
                                                  const __bf16* __restrict__ wiT,
                                                  const float* __restrict__ bias,
                                                  float* __restrict__ Z) {
  int lane = threadIdx.x & 31;
  int w = (blockIdx.x * blockDim.x + threadIdx.x) >> 5;
  int m_tile = w >> 6;            // 0..2047
  int nq     = w & 63;            // 0..63
  int m0 = m_tile << 4;
  int n0 = nq << 6;

  v8f acc[4] = {};
  for (int k = 0; k < CIN; k += 32) {
    v16bf a = load_a_frag(xb, CIN, m0, k, lane);
#pragma unroll
    for (int g = 0; g < 4; ++g) {
      v16bf bfr = load_b_frag(wiT, CIN, n0 + 16 * g, k, lane);
      acc[g] = __builtin_amdgcn_wmma_f32_16x16x32_bf16(
          false, a, false, bfr, (short)0, acc[g], false, false);
    }
  }

  int rbase = m0 + ((lane >> 4) << 3);
  int cl = lane & 15;
#pragma unroll
  for (int g = 0; g < 4; ++g) {
    int c = n0 + 16 * g + cl;
    float bv = bias[c];
#pragma unroll
    for (int v = 0; v < 8; ++v)
      Z[(size_t)(rbase + v) * FOURH + c] = acc[g][v] + bv;
  }
}

// ---- fused recurrent step: acc = Z_t + h_sel @ Wh, gates, routed state update ----
// Block = 8 waves sharing one 16-wide n-tile; the 4 gate B-panels (4 KB/K-step)
// are staged into LDS with double-buffered async copies and reused by all 8
// waves (8x cut in Wh L2 traffic). Each wave owns a 16x16 (m,n) tile in all
// four gate sections so gates + routing are wave-local.
__global__ __launch_bounds__(256) void k_lstm_step(
    const float* __restrict__ Z, const __bf16* __restrict__ whT,
    const int* __restrict__ done, const int* __restrict__ mainm,
    const float* __restrict__ c1_old, const float* __restrict__ c2_old,
    const __bf16* __restrict__ h1_old, const __bf16* __restrict__ h2_old,
    float* __restrict__ c1_new, float* __restrict__ c2_new,
    __bf16* __restrict__ h1_new, __bf16* __restrict__ h2_new,
    float* __restrict__ out, int t) {
  __shared__ __bf16 sB[2][4][16][32];   // [buf][gate][n][k]  (8 KB)

  int tid  = threadIdx.x;
  int lane = tid & 31;
  int wave = tid >> 5;                  // 0..7
  int n_tile = blockIdx.x >> 1;         // 0..63
  int m_tile = ((blockIdx.x & 1) << 3) + wave;  // 0..15
  int m0 = m_tile << 4;
  int n0 = n_tile << 4;
  const int tB = t * BATCH;
  const float* Zt = Z + (size_t)tB * FOURH;

  // async staging map: each thread moves 16 B (8 halves) per K-stage.
  int sg    = tid >> 6;                 // gate 0..3
  int sub   = tid & 63;
  int srow  = sub >> 2;                 // column n 0..15
  int schnk = (sub & 3) << 3;           // half-offset within the 32-wide K slab
  const __bf16* gsrc = whT + ((size_t)(sg * HDIM) + n0 + srow) * HDIM + schnk;
  __bf16* lslot[2] = { &sB[0][sg][srow][schnk], &sB[1][sg][srow][schnk] };

  // prologue: stage K-slab 0 into buffer 0 while we fetch Z.
  async_copy_b128(gsrc, lslot[0]);

  int rbase = m0 + ((lane >> 4) << 3);
  int cl = lane & 15;

  // init accumulators from precomputed x@Wi + b
  v8f acc[4];
#pragma unroll
  for (int g = 0; g < 4; ++g) {
#pragma unroll
    for (int v = 0; v < 8; ++v)
      acc[g][v] = Zt[(size_t)(rbase + v) * FOURH + (g * HDIM + n0 + cl)];
  }

  // h_sel @ Wh : per-lane routed A rows (main picks track 1 or 2)
  int ar = m0 + (lane & 15);
  const __bf16* hb = (mainm[tB + ar] != 0) ? h1_old : h2_old;

  int buf = 0;
  for (int k = 0; k < HDIM; k += 32) {
    if (k + 32 < HDIM) {
      async_copy_b128(gsrc + (k + 32), lslot[buf ^ 1]);
      wait_async_le1();                 // slab k has landed
    } else {
      wait_async_le0();
    }
    __syncthreads();                    // everyone's slab-k data visible

    v16bf a = load_a_frag(hb, HDIM, m0, k, lane);
#pragma unroll
    for (int g = 0; g < 4; ++g) {
      v16bf bfr = load_b_lds(&sB[buf][g][0][0], lane);
      acc[g] = __builtin_amdgcn_wmma_f32_16x16x32_bf16(
          false, a, false, bfr, (short)0, acc[g], false, false);
    }
    __syncthreads();                    // reads done before buf is overwritten
    buf ^= 1;
  }

  // gates + routed/reset state update (double-buffered, race-free)
#pragma unroll
  for (int v = 0; v < 8; ++v) {
    int r = rbase + v;
    int c = n0 + cl;
    bool mn = mainm[tB + r] != 0;
    bool dn = done[tB + r] != 0;
    size_t si = (size_t)r * HDIM + c;

    float cprev = mn ? c1_old[si] : c2_old[si];
    float zi = acc[0][v], zf = acc[1][v], zg = acc[2][v], zo = acc[3][v];
    float nc = fast_sigmoid(zf) * cprev + fast_sigmoid(zi) * fast_tanh(zg);
    float nh = fast_sigmoid(zo) * fast_tanh(nc);

    out[(size_t)(tB + r) * HDIM + c] = nh;

    float o1c = c1_old[si], o2c = c2_old[si];
    float o1h = (float)h1_old[si], o2h = (float)h2_old[si];
    float n1c = dn ? 0.f : (mn ? nc : o1c);
    float n2c = dn ? 0.f : (mn ? o2c : nc);
    float n1h = dn ? 0.f : (mn ? nh : o1h);
    float n2h = dn ? 0.f : (mn ? o2h : nh);
    c1_new[si] = n1c;
    c2_new[si] = n2c;
    h1_new[si] = (__bf16)n1h;
    h2_new[si] = (__bf16)n2h;
  }
}

extern "C" void kernel_launch(void* const* d_in, const int* in_sizes, int n_in,
                              void* d_out, int out_size, void* d_ws, size_t ws_size,
                              hipStream_t stream) {
  const float* x   = (const float*)d_in[0];
  const float* c1  = (const float*)d_in[1];
  const float* h1  = (const float*)d_in[2];
  const float* c2  = (const float*)d_in[3];
  const float* h2  = (const float*)d_in[4];
  const float* Wi  = (const float*)d_in[5];
  const float* Wh  = (const float*)d_in[6];
  const float* b   = (const float*)d_in[7];
  const int*   dn  = (const int*)d_in[8];
  const int*   mn  = (const int*)d_in[9];
  float* out = (float*)d_out;

  char* ws = (char*)d_ws;
  size_t off = 0;
  auto alloc = [&](size_t bytes) {
    char* p = ws + off;
    off += (bytes + 255) & ~(size_t)255;
    return p;
  };
  const size_t NX = (size_t)T_STEPS * BATCH * CIN;        // 33,554,432
  const size_t NW = (size_t)CIN * FOURH;                  //  4,194,304
  const size_t NS = (size_t)BATCH * HDIM;                 //    262,144
  const size_t NZ = (size_t)T_STEPS * BATCH * FOURH;      // 134,217,728

  __bf16* xb   = (__bf16*)alloc(NX * 2);                  //  64 MB
  __bf16* wiT  = (__bf16*)alloc(NW * 2);                  //   8 MB
  __bf16* whT  = (__bf16*)alloc(NW * 2);                  //   8 MB
  float*  Z    = (float*) alloc(NZ * 4);                  // 512 MB
  float*  c1s[2] = { (float*)alloc(NS * 4), (float*)alloc(NS * 4) };
  float*  c2s[2] = { (float*)alloc(NS * 4), (float*)alloc(NS * 4) };
  __bf16* h1b[2] = { (__bf16*)alloc(NS * 2), (__bf16*)alloc(NS * 2) };
  __bf16* h2b[2] = { (__bf16*)alloc(NS * 2), (__bf16*)alloc(NS * 2) };
  (void)ws_size; (void)in_sizes; (void)n_in; (void)out_size;

  // 1) precision prep (one-time)
  k_cast_bf16<<<(int)(NX / 256), 256, 0, stream>>>(x, xb, (int)NX);
  k_transpose_bf16<<<(int)(NW / 256), 256, 0, stream>>>(Wi, wiT, CIN, FOURH);
  k_transpose_bf16<<<(int)(NW / 256), 256, 0, stream>>>(Wh, whT, HDIM, FOURH);
  k_init_state<<<(int)(NS / 256), 256, 0, stream>>>(c1, h1, c2, h2,
                                                    c1s[0], c2s[0], h1b[0], h2b[0],
                                                    (int)NS);

  // 2) hoisted input projection: Z = X @ Wi + b  (HBM-bound, WMMA bf16)
  {
    // waves = (32768/16) m-tiles * (4096/64) n-quads = 131072 -> 16384 blocks
    int blocks = (int)((2048ull * 64ull * 32ull) / 256ull);
    k_gemm_xwi<<<blocks, 256, 0, stream>>>(xb, wiT, b, Z);
  }

  // 3) sequential recurrence: 128 fused step kernels (Wh stays L2-resident,
  //    B-panels staged through LDS with async copies, shared by 8 waves)
  for (int t = 0; t < T_STEPS; ++t) {
    int cur = t & 1, nxt = cur ^ 1;
    // blocks: 64 n-tiles x 2 m-halves; 8 waves (m-tiles) per block
    k_lstm_step<<<128, 256, 0, stream>>>(Z, whT, dn, mn,
                                         c1s[cur], c2s[cur], h1b[cur], h2b[cur],
                                         c1s[nxt], c2s[nxt], h1b[nxt], h2b[nxt],
                                         out, t);
  }
}
